// GNN_auto_6545530159667
// MI455X (gfx1250) — compile-verified, matrix operand order checked by reading, not run on hardware
//
#include <hip/hip_runtime.h>
#include <hip/hip_bf16.h>

typedef __attribute__((ext_vector_type(16))) _Float16 v16h;
typedef __attribute__((ext_vector_type(8)))  float    v8f;

__device__ __forceinline__ v8f wmma16(v16h a, v16h b, v8f c) {
  // D = A(16x32 f16) * B(32x16 f16) + C(16x16 f32)
  return __builtin_amdgcn_wmma_f32_16x16x32_f16(false, a, false, b, (short)0, c, false, false);
}

__device__ __forceinline__ float sigmoidf_(float x) { return 1.0f / (1.0f + __expf(-x)); }

// ---- A-matrix 16x32 f16 layout (ISA 7.12.2): lane m=L&15; lanes>=16 shift K by 8.
// Full tiles only: no row clamping, EXEC stays all-ones.
__device__ __forceinline__ v16h load_A_tile(const float* __restrict__ base, int row0,
                                            int kc, int lane) {
  int m = lane & 15;
  int koff = (lane >> 4) << 3;
  const float* r = base + (long)(row0 + m) * 64 + kc * 32;
  v16h a;
#pragma unroll
  for (int h = 0; h < 8; ++h) a[h] = (_Float16)r[koff + h];
#pragma unroll
  for (int h = 0; h < 8; ++h) a[8 + h] = (_Float16)r[16 + koff + h];
  return a;
}

// B tile from row-major W[k][n] (ld = row length). Column n = n0 + (L&15).
__device__ __forceinline__ v16h load_B_rm(const float* __restrict__ W, int ld,
                                          int kc, int n0, int lane) {
  int n = n0 + (lane & 15);
  int koff = (lane >> 4) << 3;
  v16h b;
#pragma unroll
  for (int h = 0; h < 8; ++h) b[h] = (_Float16)W[(kc * 32 + koff + h) * ld + n];
#pragma unroll
  for (int h = 0; h < 8; ++h) b[8 + h] = (_Float16)W[(kc * 32 + 16 + koff + h) * ld + n];
  return b;
}

// B tile where B = W^T : B[k][n] = W[rowbase+n][k]  (W row-major, row length ld)
__device__ __forceinline__ v16h load_B_tr(const float* __restrict__ W, int ld,
                                          int kc, int rowbase, int lane) {
  int n = lane & 15;
  int koff = (lane >> 4) << 3;
  const float* r = W + (long)(rowbase + n) * ld + kc * 32;
  v16h b;
#pragma unroll
  for (int h = 0; h < 8; ++h) b[h] = (_Float16)r[koff + h];
#pragma unroll
  for (int h = 0; h < 8; ++h) b[8 + h] = (_Float16)r[16 + koff + h];
  return b;
}

// -------------------- utility kernels --------------------
__global__ void k_zero(float4* __restrict__ p, long n4) {
  long i = (long)blockIdx.x * blockDim.x + threadIdx.x;
  long s = (long)gridDim.x * blockDim.x;
  for (; i < n4; i += s) p[i] = make_float4(0.f, 0.f, 0.f, 0.f);
}

__global__ void k_seed(float* __restrict__ hidden, const int* __restrict__ qsub,
                       const int* __restrict__ qrel, const float* __restrict__ qre) {
  int b = blockIdx.x, t = threadIdx.x;
  hidden[(long)qsub[b] * 64 + t] = qre[(long)qrel[b] * 64 + t];
}

// relWr[r][a] = rela_l[r] @ Wr ; qrW[b][a] = rela_l[qrel[b]] @ Wqr + bqr
__global__ void k_relproj(const float* __restrict__ rela_l, const float* __restrict__ Wr,
                          const float* __restrict__ Wqr, const float* __restrict__ bqr,
                          const int* __restrict__ qrel, float* __restrict__ relWr,
                          float* __restrict__ qrW, int nrel, int nb) {
  __shared__ float emb[64];
  int a = threadIdx.x;
  int bx = blockIdx.x;
  if (bx < nrel) {
    emb[a] = rela_l[(long)bx * 64 + a];
    __syncthreads();
    float acc = 0.f;
#pragma unroll
    for (int h = 0; h < 64; ++h) acc += emb[h] * Wr[h * 64 + a];
    relWr[(long)bx * 64 + a] = acc;
  } else {
    int b = bx - nrel;
    if (b < nb) {
      emb[a] = rela_l[(long)qrel[b] * 64 + a];
      __syncthreads();
      float acc = bqr[a];
#pragma unroll
      for (int h = 0; h < 64; ++h) acc += emb[h] * Wqr[h * 64 + a];
      qrW[(long)b * 64 + a] = acc;
    }
  }
}

// -------------------- nodeWs = hidden @ Ws  (WMMA, full 16-row tiles only) --------------------
__global__ __launch_bounds__(128) void k_gemm_nodeWs(const float* __restrict__ hidden,
                                                     const float* __restrict__ Ws,
                                                     float* __restrict__ out,
                                                     int ntiles) {
  int lane = threadIdx.x & 31, wid = threadIdx.x >> 5;
  int wpb = blockDim.x >> 5;
  int nloc = lane & 15, mb = (lane >> 4) * 8;
  // loop-invariant B tiles (Ws), converted once per wave
  v16h bs[4][2];
#pragma unroll
  for (int nt = 0; nt < 4; ++nt) {
    bs[nt][0] = load_B_rm(Ws, 64, 0, nt * 16, lane);
    bs[nt][1] = load_B_rm(Ws, 64, 1, nt * 16, lane);
  }
  for (int t = blockIdx.x * wpb + wid; t < ntiles; t += gridDim.x * wpb) {
    int row0 = t * 16;
    v16h a0 = load_A_tile(hidden, row0, 0, lane);
    v16h a1 = load_A_tile(hidden, row0, 1, lane);
#pragma unroll
    for (int nt = 0; nt < 4; ++nt) {
      v8f c = {};
      c = wmma16(a0, bs[nt][0], c);
      c = wmma16(a1, bs[nt][1], c);
      float* op = out + (long)(row0 + mb) * 64 + nt * 16 + nloc;
#pragma unroll
      for (int r2 = 0; r2 < 8; ++r2) op[(long)r2 * 64] = c[r2];
    }
  }
}

// scalar tail for the last N%16 rows of nodeWs
__global__ void k_tail_nodeWs(const float* __restrict__ hidden, const float* __restrict__ Ws,
                              float* __restrict__ out, int rowStart, int nRows) {
  int rIdx = blockIdx.x;
  if (rIdx >= nRows) return;
  int row = rowStart + rIdx;
  int a = threadIdx.x;  // 64
  __shared__ float hrow[64];
  hrow[a] = hidden[(long)row * 64 + a];
  __syncthreads();
  float acc = 0.f;
#pragma unroll
  for (int h = 0; h < 64; ++h) acc += hrow[h] * Ws[h * 64 + a];
  out[(long)row * 64 + a] = acc;
}

// -------------------- edge phase: gather -> attention -> scatter-add --------------------
__global__ __launch_bounds__(256) void k_edge(const int* __restrict__ sub, const int* __restrict__ rel,
                                              const int* __restrict__ obj, const int* __restrict__ eb,
                                              const float* __restrict__ hidden, const float* __restrict__ nodeWs,
                                              const float* __restrict__ relWr, const float* __restrict__ qrW,
                                              const float* __restrict__ rela_l,
                                              const float* __restrict__ w_alpha_l,
                                              const float* __restrict__ b_alpha_l,
                                              float* __restrict__ agg, int E) {
  int lane = threadIdx.x & 31, wid = threadIdx.x >> 5;
  int wpb = blockDim.x >> 5;
  int c = lane * 2;
  float wax = w_alpha_l[c], way = w_alpha_l[c + 1];
  float bav = b_alpha_l[0];
  long stride = (long)gridDim.x * wpb;
  for (long e = (long)blockIdx.x * wpb + wid; e < E; e += stride) {
    if (e + stride < E) {  // prefetch next iteration's index data (global_prefetch_b8)
      __builtin_prefetch(sub + e + stride, 0, 0);
      __builtin_prefetch(obj + e + stride, 0, 0);
    }
    int s = sub[e], r = rel[e], o = obj[e], b = eb[e];
    float2 nw = *(const float2*)(nodeWs + (long)s * 64 + c);
    float2 rw = *(const float2*)(relWr + (long)r * 64 + c);
    float2 qw = *(const float2*)(qrW + (long)b * 64 + c);
    float px = fmaxf(nw.x + rw.x + qw.x, 0.f);
    float py = fmaxf(nw.y + rw.y + qw.y, 0.f);
    float part = px * wax + py * way;
#pragma unroll
    for (int m = 1; m < 32; m <<= 1) part += __shfl_xor(part, m, 32);
    float alpha = sigmoidf_(part + bav);
    float2 hs = *(const float2*)(hidden + (long)s * 64 + c);
    float2 hr = *(const float2*)(rela_l + (long)r * 64 + c);
    float* ao = agg + (long)o * 64 + c;
    unsafeAtomicAdd(ao, alpha * hs.x * hr.x);
    unsafeAtomicAdd(ao + 1, alpha * hs.y * hr.y);
  }
}

// -------------------- fused: hidden1 = relu(agg@W_h); mask; GRU step (all WMMA) --------------------
__global__ __launch_bounds__(128) void k_node_gru(const float* __restrict__ agg,
                                                  float* __restrict__ h0,
                                                  const float* __restrict__ Wh,
                                                  const float* __restrict__ Wih,
                                                  const float* __restrict__ Whh,
                                                  const float* __restrict__ bih,
                                                  const float* __restrict__ bhh,
                                                  float* __restrict__ hidden,
                                                  int ntiles) {
  __shared__ float S[4][16][68];  // per-wave 16x64 relayout scratch (padded)
  int lane = threadIdx.x & 31, wid = threadIdx.x >> 5;
  int wpb = blockDim.x >> 5;
  int nloc = lane & 15, mb = (lane >> 4) * 8;
  float(*T)[68] = S[wid];
  // loop-invariant W_h B tiles
  v16h bh[4][2];
#pragma unroll
  for (int nt = 0; nt < 4; ++nt) {
    bh[nt][0] = load_B_rm(Wh, 64, 0, nt * 16, lane);
    bh[nt][1] = load_B_rm(Wh, 64, 1, nt * 16, lane);
  }
  for (int t = blockIdx.x * wpb + wid; t < ntiles; t += gridDim.x * wpb) {
    int row0 = t * 16;
    v16h a0 = load_A_tile(agg, row0, 0, lane);
    v16h a1 = load_A_tile(agg, row0, 1, lane);
    float rs[8];
#pragma unroll
    for (int r2 = 0; r2 < 8; ++r2) rs[r2] = 0.f;
#pragma unroll
    for (int nt = 0; nt < 4; ++nt) {
      v8f c = {};
      c = wmma16(a0, bh[nt][0], c);
      c = wmma16(a1, bh[nt][1], c);
#pragma unroll
      for (int r2 = 0; r2 < 8; ++r2) {
        float v = fmaxf(c[r2], 0.f);
        T[mb + r2][nt * 16 + nloc] = v;   // wave-private LDS, DS ops in-order per wave
        rs[r2] += v;
      }
    }
    // act_signal mask: rowsum over 16 lanes of the half-wave group
#pragma unroll
    for (int r2 = 0; r2 < 8; ++r2) {
      float v = rs[r2];
      v += __shfl_xor(v, 1, 32);
      v += __shfl_xor(v, 2, 32);
      v += __shfl_xor(v, 4, 32);
      v += __shfl_xor(v, 8, 32);
      rs[r2] = (v == 0.f) ? 0.f : 1.f;
    }
    // reload hidden1 from LDS into WMMA A layout (f16)
    int m = lane & 15;
    int koff = (lane >> 4) << 3;
    v16h x0, x1;
#pragma unroll
    for (int h = 0; h < 8; ++h) {
      x0[h]     = (_Float16)T[m][koff + h];
      x0[8 + h] = (_Float16)T[m][16 + koff + h];
      x1[h]     = (_Float16)T[m][32 + koff + h];
      x1[8 + h] = (_Float16)T[m][48 + koff + h];
    }
    v16h g0 = load_A_tile(h0, row0, 0, lane);
    v16h g1 = load_A_tile(h0, row0, 1, lane);
#pragma unroll
    for (int j = 0; j < 4; ++j) {
      int n0 = j * 16;
      v8f ir = {}, iz = {}, inn = {}, hrp = {}, hzp = {}, hnp = {};
      // gi = hidden1 @ Wih^T   (Wih is (3H,H) row-major -> B^T tiles are contiguous rows)
      ir  = wmma16(x0, load_B_tr(Wih, 64, 0, 0 + n0, lane), ir);
      ir  = wmma16(x1, load_B_tr(Wih, 64, 1, 0 + n0, lane), ir);
      iz  = wmma16(x0, load_B_tr(Wih, 64, 0, 64 + n0, lane), iz);
      iz  = wmma16(x1, load_B_tr(Wih, 64, 1, 64 + n0, lane), iz);
      inn = wmma16(x0, load_B_tr(Wih, 64, 0, 128 + n0, lane), inn);
      inn = wmma16(x1, load_B_tr(Wih, 64, 1, 128 + n0, lane), inn);
      // gh = h0 @ Whh^T
      hrp = wmma16(g0, load_B_tr(Whh, 64, 0, 0 + n0, lane), hrp);
      hrp = wmma16(g1, load_B_tr(Whh, 64, 1, 0 + n0, lane), hrp);
      hzp = wmma16(g0, load_B_tr(Whh, 64, 0, 64 + n0, lane), hzp);
      hzp = wmma16(g1, load_B_tr(Whh, 64, 1, 64 + n0, lane), hzp);
      hnp = wmma16(g0, load_B_tr(Whh, 64, 0, 128 + n0, lane), hnp);
      hnp = wmma16(g1, load_B_tr(Whh, 64, 1, 128 + n0, lane), hnp);

      int col = n0 + nloc;
      float bir = bih[col], biz = bih[64 + col], bin = bih[128 + col];
      float bhr = bhh[col], bhz = bhh[64 + col], bhn = bhh[128 + col];
      float* hp = hidden + (long)(row0 + mb) * 64 + col;
      float* gp = h0 + (long)(row0 + mb) * 64 + col;
#pragma unroll
      for (int r2 = 0; r2 < 8; ++r2) {
        float h0v = gp[(long)r2 * 64];
        float rg = sigmoidf_(ir[r2] + bir + hrp[r2] + bhr);
        float zg = sigmoidf_(iz[r2] + biz + hzp[r2] + bhz);
        float ng = tanhf(inn[r2] + bin + rg * (hnp[r2] + bhn));
        float hnew = ((1.f - zg) * ng + zg * h0v) * rs[r2];
        hp[(long)r2 * 64] = hnew;
        gp[(long)r2 * 64] = hnew;
      }
    }
  }
}

// scalar tail for the last N%16 rows of the GNN+GRU node update
__global__ void k_tail_gru(const float* __restrict__ agg, float* __restrict__ h0,
                           const float* __restrict__ Wh, const float* __restrict__ Wih,
                           const float* __restrict__ Whh, const float* __restrict__ bih,
                           const float* __restrict__ bhh, float* __restrict__ hidden,
                           int rowStart, int nRows) {
  int rIdx = blockIdx.x;
  if (rIdx >= nRows) return;
  int row = rowStart + rIdx;
  int a = threadIdx.x;  // 64
  __shared__ float arow[64], h1[64], h0row[64];
  __shared__ float msk;
  arow[a] = agg[(long)row * 64 + a];
  h0row[a] = h0[(long)row * 64 + a];
  __syncthreads();
  float acc = 0.f;
#pragma unroll
  for (int h = 0; h < 64; ++h) acc += arow[h] * Wh[h * 64 + a];
  h1[a] = fmaxf(acc, 0.f);
  __syncthreads();
  if (a == 0) {
    float s = 0.f;
#pragma unroll
    for (int h = 0; h < 64; ++h) s += h1[h];
    msk = (s == 0.f) ? 0.f : 1.f;
  }
  __syncthreads();
  float ir = bih[a], iz = bih[64 + a], inn = bih[128 + a];
  float hr = bhh[a], hz = bhh[64 + a], hn = bhh[128 + a];
#pragma unroll
  for (int h = 0; h < 64; ++h) {
    float x = h1[h], g = h0row[h];
    ir += x * Wih[a * 64 + h];
    iz += x * Wih[(64 + a) * 64 + h];
    inn += x * Wih[(128 + a) * 64 + h];
    hr += g * Whh[a * 64 + h];
    hz += g * Whh[(64 + a) * 64 + h];
    hn += g * Whh[(128 + a) * 64 + h];
  }
  float rg = sigmoidf_(ir + hr);
  float zg = sigmoidf_(iz + hz);
  float ng = tanhf(inn + rg * hn);
  float hnew = ((1.f - zg) * ng + zg * h0row[a]) * msk;
  hidden[(long)row * 64 + a] = hnew;
  h0[(long)row * 64 + a] = hnew;
}

// -------------------- readout: scores = hidden @ W_final --------------------
__global__ __launch_bounds__(256) void k_readout(const float* __restrict__ hidden,
                                                 const float* __restrict__ Wf,
                                                 float* __restrict__ out, int N) {
  int lane = threadIdx.x & 31, wid = threadIdx.x >> 5;
  int wpb = blockDim.x >> 5;
  float w0 = Wf[2 * lane], w1 = Wf[2 * lane + 1];
  for (int n = blockIdx.x * wpb + wid; n < N; n += gridDim.x * wpb) {
    float2 h = *(const float2*)(hidden + (long)n * 64 + 2 * lane);
    float p = h.x * w0 + h.y * w1;
#pragma unroll
    for (int m = 1; m < 32; m <<= 1) p += __shfl_xor(p, m, 32);
    if (lane == 0) out[n] = p;
  }
}

extern "C" void kernel_launch(void* const* d_in, const int* in_sizes, int n_in,
                              void* d_out, int out_size, void* d_ws, size_t ws_size,
                              hipStream_t stream) {
  const int* sub = (const int*)d_in[0];
  const int* rel = (const int*)d_in[1];
  const int* obj = (const int*)d_in[2];
  const int* eb = (const int*)d_in[3];
  const int* qrel = (const int*)d_in[4];
  const int* qsub = (const int*)d_in[5];
  const float* rela = (const float*)d_in[6];
  const float* qre = (const float*)d_in[7];
  const float* Ws = (const float*)d_in[8];
  const float* Wr = (const float*)d_in[9];
  const float* Wqr = (const float*)d_in[10];
  const float* bqr = (const float*)d_in[11];
  const float* wal = (const float*)d_in[12];
  const float* bal = (const float*)d_in[13];
  const float* Wh = (const float*)d_in[14];
  const float* Wih = (const float*)d_in[15];
  const float* Whh = (const float*)d_in[16];
  const float* bih = (const float*)d_in[17];
  const float* bhh = (const float*)d_in[18];
  const float* Wf = (const float*)d_in[19];

  int E = in_sizes[0];
  int Bq = in_sizes[4];
  int NREL = in_sizes[7] / 64;       // (NREL2, H=64)
  int Lnum = in_sizes[8] / 4096;     // (L, 64, 64)
  int N = out_size;                  // scores per node
  long NH = (long)N * 64;

  float* ws = (float*)d_ws;
  float* hidden = ws;
  float* h0 = ws + NH;
  float* agg = ws + 2 * NH;
  float* nodeWs = ws + 3 * NH;
  float* relWr = ws + 4 * NH;
  float* qrW = relWr + (long)NREL * 64;

  {  // zero hidden + h0 (contiguous)
    long n4 = (2 * NH) / 4;
    int blocks = (int)((n4 + 255) / 256);
    if (blocks > 4096) blocks = 4096;
    k_zero<<<blocks, 256, 0, stream>>>((float4*)hidden, n4);
  }
  k_seed<<<Bq, 64, 0, stream>>>(hidden, qsub, qrel, qre);

  int ntFull = N / 16;              // full 16-row tiles for WMMA kernels
  int rem = N - ntFull * 16;        // scalar tail rows
  int rowStart = ntFull * 16;
  int gemmBlocks = (ntFull + 3) / 4;
  if (gemmBlocks > 8192) gemmBlocks = 8192;
  if (gemmBlocks < 1) gemmBlocks = 1;
  int edgeBlocks = (E + 7) / 8;
  if (edgeBlocks > 16384) edgeBlocks = 16384;

  for (int l = 0; l < Lnum; ++l) {
    const float* rela_l = rela + (long)l * NREL * 64;
    k_relproj<<<NREL + Bq, 64, 0, stream>>>(rela_l, Wr + l * 4096, Wqr + l * 4096,
                                            bqr + l * 64, qrel, relWr, qrW, NREL, Bq);
    if (ntFull > 0)
      k_gemm_nodeWs<<<gemmBlocks, 128, 0, stream>>>(hidden, Ws + l * 4096, nodeWs, ntFull);
    if (rem > 0)
      k_tail_nodeWs<<<rem, 64, 0, stream>>>(hidden, Ws + l * 4096, nodeWs, rowStart, rem);
    {
      long n4 = NH / 4;
      int blocks = (int)((n4 + 255) / 256);
      if (blocks > 4096) blocks = 4096;
      k_zero<<<blocks, 256, 0, stream>>>((float4*)agg, n4);
    }
    k_edge<<<edgeBlocks, 256, 0, stream>>>(sub, rel, obj, eb, hidden, nodeWs, relWr, qrW,
                                           rela_l, wal + l * 64, bal + l, agg, E);
    if (ntFull > 0)
      k_node_gru<<<gemmBlocks, 128, 0, stream>>>(agg, h0, Wh + l * 4096, Wih + (long)l * 12288,
                                                 Whh + (long)l * 12288, bih + l * 192,
                                                 bhh + l * 192, hidden, ntFull);
    if (rem > 0)
      k_tail_gru<<<rem, 64, 0, stream>>>(agg, h0, Wh + l * 4096, Wih + (long)l * 12288,
                                         Whh + (long)l * 12288, bih + l * 192,
                                         bhh + l * 192, hidden, rowStart, rem);
  }

  int roBlocks = (N + 7) / 8;
  if (roBlocks > 8192) roBlocks = 8192;
  k_readout<<<roBlocks, 256, 0, stream>>>(hidden, Wf, (float*)d_out, N);
}
